// DecoderMLP_65335042506961
// MI455X (gfx1250) — matrix operand order repaired and checked by reading, not compiled, not run on hardware
//
#include <hip/hip_runtime.h>

typedef __attribute__((ext_vector_type(16))) _Float16 v16h;
typedef __attribute__((ext_vector_type(8)))  float    v8f;

#define NN 20000
#define CC 256
#define SS 8

// ---------------------------------------------------------------------------
// helpers
// ---------------------------------------------------------------------------
__device__ inline v8f wmma16(v16h a, v16h b, v8f c) {
  // D = A(16x32 f16) * B(32x16 f16) + C(16x16 f32)
  return __builtin_amdgcn_wmma_f32_16x16x32_f16(
      false, a, false, b, (short)0, c, false, false);
}

// LDS tile: 64 rows x 256 f16, row = 512B = 32 chunks of 16B.
// chunk (m, c) stored at uint4 index  m*32 + (c ^ (m & 15))   (bank-conflict-free swizzle)

// A fragment (16x32): lane l -> row m = base + (l&15);
//  l<16: K {k0..k0+15} ; l>=16: K {k0+16..k0+31}
__device__ inline v16h loadA(const uint4* L, int rb, int k0, int lane) {
  int m  = rb + (lane & 15);
  int hi = (lane >> 4) & 1;
  int cb = k0 >> 3;
  union { v16h v; uint4 q[2]; } f;
  f.q[0] = L[m * 32 + ((cb + 2 * hi)     ^ (m & 15))];
  f.q[1] = L[m * 32 + ((cb + 2 * hi + 1) ^ (m & 15))];
  return f.v;
}
// A fragment where only 8 rows exist (rows replicated into 8..15, results ignored)
__device__ inline v16h loadA8(const uint4* L, int rb, int k0, int lane) {
  int m  = rb + (lane & 7);
  int hi = (lane >> 4) & 1;
  int cb = k0 >> 3;
  union { v16h v; uint4 q[2]; } f;
  f.q[0] = L[m * 32 + ((cb + 2 * hi)     ^ (m & 15))];
  f.q[1] = L[m * 32 + ((cb + 2 * hi + 1) ^ (m & 15))];
  return f.v;
}
// B fragment (32x16) from an LDS tile whose rows are the B columns:
// lane l -> col n = base + (l&7) (8 valid rows), K contiguous 16 at k0 + 16*(l>=16)
__device__ inline v16h loadB8(const uint4* L, int rb, int k0, int lane) {
  int n  = rb + (lane & 7);
  int hi = (lane >> 4) & 1;
  int cb = (k0 >> 3) + hi * 2;
  union { v16h v; uint4 q[2]; } f;
  f.q[0] = L[n * 32 + ((cb)     ^ (n & 15))];
  f.q[1] = L[n * 32 + ((cb + 1) ^ (n & 15))];
  return f.v;
}
// B fragment from pre-transposed global weights WT[n][k] (f16):
__device__ inline v16h loadBg(const unsigned short* WT, int n0, int k0, int lane) {
  int n = n0 + (lane & 15);
  int k = k0 + ((lane >> 4) & 1) * 16;
  const uint4* p = (const uint4*)(WT + (size_t)n * CC + k);
  union { v16h v; uint4 q[2]; } f;
  f.q[0] = p[0];
  f.q[1] = p[1];
  return f.v;
}

__device__ inline unsigned short h2u(_Float16 h) {
  return __builtin_bit_cast(unsigned short, h);
}

// one 256x256 hidden layer, tile [64 rows x 256 cols], computed in place in LDS
// 16 waves: wm = wid&3 (16-row group), wn = wid>>2 (64-col group, 4 tiles)
__device__ inline void layer_inplace(uint4* L4, unsigned short* Ls,
                                     const unsigned short* WT, const float* bias,
                                     bool relu, int lane, int wm, int wn) {
  v8f acc[4] = {};
  for (int k0 = 0; k0 < CC; k0 += 32) {
    v16h a = loadA(L4, wm * 16, k0, lane);
#pragma unroll
    for (int t = 0; t < 4; ++t) {
      v16h b = loadBg(WT, wn * 64 + t * 16, k0, lane);
      acc[t] = wmma16(a, b, acc[t]);
    }
  }
  __syncthreads();  // all reads done before any in-place write
#pragma unroll
  for (int t = 0; t < 4; ++t) {
    int n = wn * 64 + t * 16 + (lane & 15);
    float bv = bias[n];
#pragma unroll
    for (int r = 0; r < 8; ++r) {
      int m = wm * 16 + r + ((lane >> 4) & 1) * 8;
      float v = acc[t][r] + bv;
      if (relu) v = fmaxf(v, 0.f);
      Ls[(m * 32 + ((n >> 3) ^ (m & 15))) * 8 + (n & 7)] = h2u((_Float16)v);
    }
  }
  __syncthreads();
}

// ---------------------------------------------------------------------------
// kernel 0: weights fp32 -> f16, transposed to WT[n][k] in workspace
// ---------------------------------------------------------------------------
struct TMat { const float* src; int ncols; int npad; int dstoff; };
struct TArgs { TMat m[22]; };

__global__ void transpose_kernel(TArgs ta, unsigned short* wsh) {
  TMat mt = ta.m[blockIdx.x];
  int e = blockIdx.y * 256 + threadIdx.x;
  if (e >= mt.npad * CC) return;
  int n = e >> 8, k = e & 255;
  float v = (n < mt.ncols) ? mt.src[(size_t)k * mt.ncols + n] : 0.f;
  wsh[(size_t)mt.dstoff + e] = h2u((_Float16)v);
}

// ---------------------------------------------------------------------------
// kernel 1: head MLPs (deg / pts / cls / pos), 512 threads = 16 waves
// ---------------------------------------------------------------------------
struct HeadArgs {
  const float* bh0[4];
  const float* bo[4];
  int  wtH[4];
  int  wtO[4];
  int  cout[4];
  int  ntiles[4];
  long outOff[4];
  int  frelu[4];
};

__global__ void heads_kernel(const float* __restrict__ emb,
                             const unsigned short* __restrict__ wsh,
                             float* __restrict__ out, HeadArgs ha) {
  __shared__ __align__(16) unsigned short sA[64 * CC];
  uint4* L4 = (uint4*)sA;
  const int tid  = threadIdx.x;
  const int lane = tid & 31;
  const int wid  = tid >> 5;
  const int wm = wid & 3, wn = wid >> 2;
  const int head = blockIdx.y;
  const long rowBase = (long)blockIdx.x * 64;

  // stage X tile (fp32 -> f16, swizzled); 2048 chunks over 512 threads
  for (int j = 0; j < 4; ++j) {
    int cidx = tid + j * 512;
    int m = cidx >> 5, kc = cidx & 31;
    long rg = rowBase + m;
    union { _Float16 h[8]; uint4 q; } pk;
    if (rg < NN) {
      const float4* s4 = (const float4*)(emb + rg * CC + kc * 8);
      float4 x0 = s4[0], x1 = s4[1];
      pk.h[0] = (_Float16)x0.x; pk.h[1] = (_Float16)x0.y;
      pk.h[2] = (_Float16)x0.z; pk.h[3] = (_Float16)x0.w;
      pk.h[4] = (_Float16)x1.x; pk.h[5] = (_Float16)x1.y;
      pk.h[6] = (_Float16)x1.z; pk.h[7] = (_Float16)x1.w;
    } else {
#pragma unroll
      for (int e = 0; e < 8; ++e) pk.h[e] = (_Float16)0.f;
    }
    L4[m * 32 + (kc ^ (m & 15))] = pk.q;
  }
  __syncthreads();

  for (int layer = 0; layer < 2; ++layer)
    layer_inplace(L4, sA, wsh + ha.wtH[head] + layer * 65536,
                  ha.bh0[head] + layer * CC, true, lane, wm, wn);

  // output head (cout columns, padded tiles); N-tiles distributed over wn = 0..3
  {
    const unsigned short* WT = wsh + ha.wtO[head];
    const float* bias = ha.bo[head];
    float* outp = out + ha.outOff[head];
    int cout = ha.cout[head], nt = ha.ntiles[head];
    for (int t = wn; t < nt; t += 4) {
      v8f acc = {};
      for (int k0 = 0; k0 < CC; k0 += 32) {
        v16h a = loadA(L4, wm * 16, k0, lane);
        v16h b = loadBg(WT, t * 16, k0, lane);
        acc = wmma16(a, b, acc);
      }
      int n = t * 16 + (lane & 15);
      if (n < cout) {
        float bv = bias[n];
#pragma unroll
        for (int r = 0; r < 8; ++r) {
          int m = wm * 16 + r + ((lane >> 4) & 1) * 8;
          long rg = rowBase + m;
          if (rg < NN) {
            float v = acc[r] + bv;
            if (ha.frelu[head]) v = fmaxf(v, 0.f);
            outp[rg * cout + n] = v;
          }
        }
      }
    }
  }
}

// ---------------------------------------------------------------------------
// kernel 2: fused decoder (mean/sigma -> var -> gen MLP -> pairwise RMS errors)
// 512 threads = 16 waves, 8 nodes per block
// ---------------------------------------------------------------------------
__global__ void decoder_kernel(const float* __restrict__ emb,
                               const float* __restrict__ emb_inter,
                               const int* __restrict__ nidx,
                               const float* __restrict__ nmask,
                               const float* __restrict__ stdz,
                               const unsigned short* __restrict__ wsh,
                               int wtMean, int wtSigma, int wtGen,
                               const float* __restrict__ b_mean,
                               const float* __restrict__ b_sigma,
                               const float* __restrict__ b_gen_h,
                               const float* __restrict__ b_gen_o,
                               float* __restrict__ out_err) {
  __shared__ __align__(16) unsigned short sA[64 * CC];  // var -> gen (in place)
  __shared__ __align__(16) unsigned short sB[64 * CC];  // gathered targets
  uint4* LA = (uint4*)sA;
  uint4* LB = (uint4*)sB;
  const int tid  = threadIdx.x;
  const int lane = tid & 31;
  const int wid  = tid >> 5;
  const int wm = wid & 3, wn = wid >> 2;
  const int dec = blockIdx.y;
  const long nodeBase = (long)blockIdx.x * 8;

  const float* origin = (dec == 0) ? emb : emb_inter;          // embs[dec]
  const float* dest   = emb_inter + (size_t)dec * NN * CC;     // embs[dec+1]

  // 1) stage origin rows (8 nodes x 256) into sA rows 0..7 (256 chunks)
  if (tid < 256) {
    int m = tid >> 5, kc = tid & 31;
    const float4* s4 = (const float4*)(origin + (nodeBase + m) * CC + kc * 8);
    float4 x0 = s4[0], x1 = s4[1];
    union { _Float16 h[8]; uint4 q; } pk;
    pk.h[0] = (_Float16)x0.x; pk.h[1] = (_Float16)x0.y;
    pk.h[2] = (_Float16)x0.z; pk.h[3] = (_Float16)x0.w;
    pk.h[4] = (_Float16)x1.x; pk.h[5] = (_Float16)x1.y;
    pk.h[6] = (_Float16)x1.z; pk.h[7] = (_Float16)x1.w;
    LA[m * 32 + (kc ^ (m & 15))] = pk.q;
  }
  __syncthreads();

  // 2) mean / sigma GEMMs: wave w owns 16 cols (one tile of each)
  v8f am = {}, as_ = {};
  for (int k0 = 0; k0 < CC; k0 += 32) {
    v16h a = loadA8(LA, 0, k0, lane);
    am  = wmma16(a, loadBg(wsh + wtMean,  wid * 16, k0, lane), am);
    as_ = wmma16(a, loadBg(wsh + wtSigma, wid * 16, k0, lane), as_);
  }
  __syncthreads();  // reads of sA rows 0..7 done

  // 3) var = mean + exp(sigma) * std_z  ->  sA rows 0..63 (f16, swizzled)
  if (lane < 16) {
    int n = wid * 16 + lane;
    float bm = b_mean[n], bs = b_sigma[n];
    float sz[8];
#pragma unroll
    for (int s = 0; s < 8; ++s) sz[s] = stdz[s * CC + n];
#pragma unroll
    for (int r = 0; r < 8; ++r) {
      float mv = am[r] + bm;
      float ev = expf(as_[r] + bs);
#pragma unroll
      for (int s = 0; s < 8; ++s) {
        int row = r * 8 + s;
        float v = mv + ev * sz[s];
        sA[(row * 32 + ((n >> 3) ^ (row & 15))) * 8 + (n & 7)] = h2u((_Float16)v);
      }
    }
  }

  // 4) gather masked neighbors -> sB (2048 chunks over 512 threads)
  for (int j = 0; j < 4; ++j) {
    int cidx = tid + j * 512;
    int m = cidx >> 5, kc = cidx & 31;
    int nl = m >> 3, s = m & 7;
    long ng = nodeBase + nl;
    int   idx = nidx[ng * SS + s];
    float mk  = nmask[ng * SS + s];
    const float4* s4 = (const float4*)(dest + (size_t)idx * CC + kc * 8);
    float4 x0 = s4[0], x1 = s4[1];
    union { _Float16 h[8]; uint4 q; } pk;
    pk.h[0] = (_Float16)(x0.x * mk); pk.h[1] = (_Float16)(x0.y * mk);
    pk.h[2] = (_Float16)(x0.z * mk); pk.h[3] = (_Float16)(x0.w * mk);
    pk.h[4] = (_Float16)(x1.x * mk); pk.h[5] = (_Float16)(x1.y * mk);
    pk.h[6] = (_Float16)(x1.z * mk); pk.h[7] = (_Float16)(x1.w * mk);
    LB[m * 32 + (kc ^ (m & 15))] = pk.q;
  }
  __syncthreads();

  // 5) gen MLP: 3 relu layers + linear out, in place in sA
  for (int layer = 0; layer < 4; ++layer) {
    const unsigned short* WT = wsh + wtGen + (dec * 4 + layer) * 65536;
    const float* bias = (layer < 3) ? (b_gen_h + (dec * 3 + layer) * CC)
                                    : (b_gen_o + dec * CC);
    layer_inplace(LA, sA, WT, bias, layer < 3, lane, wm, wn);
  }

  // 6) per-node pairwise errors via WMMA Gram matrices (waves 0..7 -> node)
  if (wid < 8) {
    int rb = wid * 8;
    v8f cgg = {}, ctt = {}, cgt = {};
    for (int k0 = 0; k0 < CC; k0 += 32) {
      v16h ag = loadA8(LA, rb, k0, lane);
      v16h at = loadA8(LB, rb, k0, lane);
      v16h bg = loadB8(LA, rb, k0, lane);
      v16h bt = loadB8(LB, rb, k0, lane);
      cgg = wmma16(ag, bg, cgg);
      ctt = wmma16(at, bt, ctt);
      cgt = wmma16(ag, bt, cgt);
    }
    const float inv = 1.f / (float)CC;
    float g2[8], t2[8];
#pragma unroll
    for (int s = 0; s < 8; ++s) {
      g2[s] = __builtin_bit_cast(
                  float, __builtin_amdgcn_readlane(
                             __builtin_bit_cast(int, cgg[s]), s)) * inv;
      t2[s] = __builtin_bit_cast(
                  float, __builtin_amdgcn_readlane(
                             __builtin_bit_cast(int, ctt[s]), s)) * inv;
    }
    float t2v = 0.f;
#pragma unroll
    for (int s = 0; s < 8; ++s) t2v = (lane == s) ? t2[s] : t2v;

    if (lane < 8) {
      long base = ((long)dec * NN + (nodeBase + wid)) * 64;
#pragma unroll
      for (int r = 0; r < 8; ++r) {
        float cr = cgt[r] * inv;                     // cross[s=r][t=lane]
        float sq = fmaxf(g2[r] + t2v - 2.f * cr, 0.f);
        out_err[base + r * 8 + lane] = sqrtf(sq + 1e-12f);
      }
    }
  }
}

// ---------------------------------------------------------------------------
// launcher
// ---------------------------------------------------------------------------
extern "C" void kernel_launch(void* const* d_in, const int* in_sizes, int n_in,
                              void* d_out, int out_size, void* d_ws, size_t ws_size,
                              hipStream_t stream) {
  (void)in_sizes; (void)n_in; (void)out_size; (void)ws_size;
  const float* emb       = (const float*)d_in[0];
  const float* emb_inter = (const float*)d_in[1];
  const int*   nidx      = (const int*)d_in[2];
  const float* nmask     = (const float*)d_in[3];
  const float* stdz      = (const float*)d_in[4];
  float* out = (float*)d_out;
  unsigned short* wsh = (unsigned short*)d_ws;

  // workspace layout (f16 elements)
  const int wtHeadH0 = 0;                 // 4 heads x 2 layers x 65536
  const int wtO[4]   = {524288, 528384, 561152, 638976};
  const int wtMean   = 643072;
  const int wtSigma  = 708608;
  const int wtGen    = 774144;            // 2 dec x 4 layers x 65536

  // ---- transpose/convert descriptors ----
  TArgs ta;
  int mi = 0;
  const int wIdx[4]  = {5, 9, 13, 17};
  const int couts[4] = {1, 128, 300, 6};
  const int npads[4] = {16, 128, 304, 16};
  for (int h = 0; h < 4; ++h)
    for (int l = 0; l < 2; ++l)
      ta.m[mi++] = TMat{(const float*)d_in[wIdx[h]] + l * 65536, 256, 256,
                        wtHeadH0 + h * 131072 + l * 65536};
  for (int h = 0; h < 4; ++h)
    ta.m[mi++] = TMat{(const float*)d_in[wIdx[h] + 2], couts[h], npads[h], wtO[h]};
  ta.m[mi++] = TMat{(const float*)d_in[21], 256, 256, wtMean};
  ta.m[mi++] = TMat{(const float*)d_in[23], 256, 256, wtSigma};
  for (int dc = 0; dc < 2; ++dc)
    for (int l = 0; l < 4; ++l) {
      const float* src = (l < 3)
          ? (const float*)d_in[25] + (dc * 3 + l) * 65536
          : (const float*)d_in[27] + dc * 65536;
      ta.m[mi++] = TMat{src, 256, 256, wtGen + (dc * 4 + l) * 65536};
    }
  transpose_kernel<<<dim3(22, 304), 256, 0, stream>>>(ta, wsh);

  // ---- heads ----
  HeadArgs ha;
  const long outOffs[4] = {0, 20000, 2580000, 8580000};
  const int  ntiles[4]  = {1, 8, 19, 1};
  const int  frelu[4]   = {1, 0, 0, 0};
  for (int h = 0; h < 4; ++h) {
    ha.bh0[h]   = (const float*)d_in[wIdx[h] + 1];
    ha.bo[h]    = (const float*)d_in[wIdx[h] + 3];
    ha.wtH[h]   = wtHeadH0 + h * 131072;
    ha.wtO[h]   = wtO[h];
    ha.cout[h]  = couts[h];
    ha.ntiles[h]= ntiles[h];
    ha.outOff[h]= outOffs[h];
    ha.frelu[h] = frelu[h];
  }
  heads_kernel<<<dim3(313, 4), 512, 0, stream>>>(emb, wsh, out, ha);

  // ---- decoders ----
  float* out_err = out + 8700000;
  decoder_kernel<<<dim3(2500, 2), 512, 0, stream>>>(
      emb, emb_inter, nidx, nmask, stdz, wsh,
      wtMean, wtSigma, wtGen,
      (const float*)d_in[22], (const float*)d_in[24],
      (const float*)d_in[26], (const float*)d_in[28],
      out_err);
}
